// LConv_89756226552252
// MI455X (gfx1250) — compile-verified, matrix operand order checked by reading, not run on hardware
//
#include <hip/hip_runtime.h>

typedef __attribute__((ext_vector_type(16))) _Float16 v16h;
typedef __attribute__((ext_vector_type(8)))  _Float16 v8h;
typedef __attribute__((ext_vector_type(8)))  float    v8f;

#define CIN   128
#define COUT  128
#define HW    112
#define PLANE (HW*HW)          // 12544
#define KTOT  (3*CIN)          // 384
#define BATCH 16
#define MTOT  (BATCH*PLANE)    // 200704 pixels

// ---------------------------------------------------------------------------
// Kernel 0: transpose + convert abc (K x N fp32) -> bt (N x K f16, K contiguous)
// ---------------------------------------------------------------------------
__global__ void prep_bt(const float* __restrict__ abc, _Float16* __restrict__ bt) {
    int idx = blockIdx.x * 256 + threadIdx.x;          // 128*384 = 49152
    if (idx < COUT * KTOT) {
        int o = idx / KTOT, k = idx % KTOT;
        bt[idx] = (_Float16)abc[k * COUT + o];
    }
}

// ---------------------------------------------------------------------------
// Kernel 1: separable feature builder.
// One workgroup per (b, output row i). Produces feat[m][k] f16, K contiguous,
// m = b*12544 + i*112 + j. k: [0,128)=xpart (dy weight), [128,256)=ypart (dx
// weight), [256,384)=box.
// ---------------------------------------------------------------------------
#define SXW 120                 // 112 + 3 pad each side + 2 spare
#define FBS 392                 // fbuf row stride in halfs (784B, 16B aligned, 2-way banks)

__global__ __launch_bounds__(128) void lconv_features(
        const float* __restrict__ x, _Float16* __restrict__ feat) {
    __shared__ float sx[7 * SXW];
    __shared__ float vb[SXW];
    __shared__ float vl[SXW];
    __shared__ __align__(16) _Float16 fbuf[HW * FBS];  // 87808 B

    const int tid = threadIdx.x;
    const int i   = blockIdx.x;     // output row
    const int b   = blockIdx.y;     // batch

    for (int c = 0; c < CIN; ++c) {
        // load 7-row padded slice for channel c
        const float* xp = x + ((size_t)b * CIN + c) * PLANE;
        for (int idx = tid; idx < 7 * SXW; idx += 128) {
            int dy = idx / SXW, col = idx % SXW;
            int r = i - 3 + dy, jj = col - 3;
            float v = 0.f;
            if (r >= 0 && r < HW && jj >= 0 && jj < HW)
                v = xp[r * HW + jj];
            sx[idx] = v;
        }
        __syncthreads();

        // vertical pass: box + dy-weighted sums per padded column
        if (tid < SXW) {
            float s = 0.f, sl = 0.f;
            #pragma unroll
            for (int dy = 0; dy < 7; ++dy) {
                float v = sx[dy * SXW + tid];
                s  += v;
                sl += (float)(dy - 3) * v;
            }
            vb[tid] = s;
            vl[tid] = sl;
        }
        __syncthreads();

        // horizontal pass + stage into fbuf (K-contiguous layout)
        if (tid < HW) {
            float f0 = 0.f, f1 = 0.f, f2 = 0.f;
            #pragma unroll
            for (int dx = 0; dx < 7; ++dx) {
                float bv = vb[tid + dx];
                float lv = vl[tid + dx];
                f2 += bv;
                f1 += (float)(dx - 3) * bv;   // ypart: horizontal linear
                f0 += lv;                      // xpart: vertical linear
            }
            fbuf[tid * FBS +           c] = (_Float16)f0;
            fbuf[tid * FBS +     CIN + c] = (_Float16)f1;
            fbuf[tid * FBS + 2 * CIN + c] = (_Float16)f2;
        }
        __syncthreads();
    }

    // contiguous burst: 112 rows x 384 halfs -> global (fully coalesced b128)
    _Float16* g = feat + ((size_t)b * PLANE + (size_t)i * HW) * KTOT;
    for (int u = tid; u < HW * (KTOT / 8); u += 128) {   // 112*48 b128 units
        int row = u / (KTOT / 8), q = u % (KTOT / 8);
        uint4 v = *reinterpret_cast<const uint4*>(&fbuf[row * FBS + q * 8]);
        *reinterpret_cast<uint4*>(
            reinterpret_cast<char*>(g) + (size_t)row * (KTOT * 2) + q * 16) = v;
    }
}

// ---------------------------------------------------------------------------
// Kernel 2: WMMA GEMM.  D(o x pixel) = bt(o x k) * feat^T(k x pixel) + bias.
// Workgroup = 256 threads / 8 waves; each wave: 16 pixels x 128 COUT.
// The wave's full B-operand slice (16 pixels x K=384 f16 = 96 VGPRs) is
// preloaded into registers in one clause, so the 8 accumulator chains only
// stream the shared, L1-resident bt (A) fragments.
// Fragment layout (CDNA5 16-bit A/B):
//   lane<16 : row r0+lane,    K {k0..k0+7, k0+16..k0+23}
//   lane>=16: row r0+lane-16, K {k0+8..k0+15, k0+24..k0+31}
// i.e. two b128 loads from a K-contiguous row-major f16 buffer.
// ---------------------------------------------------------------------------
__device__ inline v16h load_frag(const _Float16* __restrict__ base,
                                 size_t r0, int k0, int lane) {
    size_t r  = r0 + (size_t)(lane & 15);
    int    kb = k0 + ((lane >> 4) << 3);
    const _Float16* p = base + r * KTOT + kb;
    v8h lo = *(const v8h*)p;          // K kb .. kb+7
    v8h hi = *(const v8h*)(p + 16);   // K kb+16 .. kb+23
    return __builtin_shufflevector(lo, hi, 0, 1, 2, 3, 4, 5, 6, 7,
                                           8, 9, 10, 11, 12, 13, 14, 15);
}

__global__ __launch_bounds__(256) void lconv_gemm(
        const _Float16* __restrict__ feat, const _Float16* __restrict__ bt,
        const float* __restrict__ bias, float* __restrict__ out) {
    const int lane = threadIdx.x & 31;
    const int wave = threadIdx.x >> 5;                 // 0..7
    const size_t p0 = (size_t)blockIdx.x * 128 + (size_t)wave * 16;  // pixel base

    // preload this wave's entire B-operand (16 pixels x 384 K) -> 12 fragments
    v16h bf[KTOT / 32];
    #pragma unroll
    for (int ks = 0; ks < KTOT / 32; ++ks)
        bf[ks] = load_frag(feat, p0, ks * 32, lane);

    v8f acc[8];
    #pragma unroll
    for (int nt = 0; nt < 8; ++nt) {
        v8f a = {0.f, 0.f, 0.f, 0.f, 0.f, 0.f, 0.f, 0.f};
        #pragma unroll
        for (int ks = 0; ks < KTOT / 32; ++ks) {
            v16h af = load_frag(bt, (size_t)nt * 16, ks * 32, lane);  // A: 16 outs (M)
            a = __builtin_amdgcn_wmma_f32_16x16x32_f16(
                false, af, false, bf[ks], (short)0, a, false, false);
        }
        acc[nt] = a;
    }

    // store: N=pixel=lane%16 -> lanes are contiguous in NCHW output
    const int bb    = (int)(p0 / PLANE);               // tiles never cross batch
    const size_t pr = p0 - (size_t)bb * PLANE + (lane & 15);
    const int half8 = (lane >> 4) << 3;
    #pragma unroll
    for (int nt = 0; nt < 8; ++nt) {
        #pragma unroll
        for (int r = 0; r < 8; ++r) {
            int o = nt * 16 + r + half8;
            float v = acc[nt][r] + bias[o];
            out[((size_t)(bb * COUT + o)) * PLANE + pr] = v;
        }
    }
}

// ---------------------------------------------------------------------------
extern "C" void kernel_launch(void* const* d_in, const int* in_sizes, int n_in,
                              void* d_out, int out_size, void* d_ws, size_t ws_size,
                              hipStream_t stream) {
    const float* x    = (const float*)d_in[0];   // (16,128,112,112)
    const float* abc  = (const float*)d_in[1];   // (384,128)
    const float* bias = (const float*)d_in[2];   // (128,)
    float*       out  = (float*)d_out;           // (16,128,112,112)

    char* ws = (char*)d_ws;
    _Float16* bt   = (_Float16*)ws;                              // 98304 B
    _Float16* feat = (_Float16*)(ws + (size_t)COUT * KTOT * 2);  // 154.1 MB

    prep_bt<<<192, 256, 0, stream>>>(abc, bt);
    lconv_features<<<dim3(HW, BATCH), 128, 0, stream>>>(x, feat);
    lconv_gemm<<<MTOT / 128, 256, 0, stream>>>(feat, bt, bias, out);
}
